// DeformConv2d_33165737460338
// MI455X (gfx1250) — matrix operand order repaired and checked by reading, not compiled, not run on hardware
//
#include <hip/hip_runtime.h>
#include <hip/hip_bf16.h>
#include <math.h>

typedef __attribute__((ext_vector_type(16))) __bf16        v16bf;
typedef __attribute__((ext_vector_type(8)))  float         v8f;
typedef __attribute__((ext_vector_type(8)))  unsigned int  v8u;
typedef __attribute__((ext_vector_type(4)))  unsigned int  v4u;

#define BATCH 8
#define CIN   64
#define HH    128
#define WW    128
#define NOFF  18
#define NTAP  9
#define HWSZ  (HH * WW)   // 16384

static __device__ __forceinline__ unsigned short f32_to_bf16(float f) {
  unsigned int u = __float_as_uint(f);
  unsigned int r = u + 0x7FFFu + ((u >> 16) & 1u);   // round-to-nearest-even
  return (unsigned short)(r >> 16);
}

// ---------------------------------------------------------------------------
// Kernel 1: offset conv (3x3, pad 1, 64 -> 18 channels), direct form.
// One thread per pixel computes all 18 offset channels; w_p staged in LDS.
// ---------------------------------------------------------------------------
__global__ void __launch_bounds__(256)
offset_conv_kernel(const float* __restrict__ x, const float* __restrict__ w_p,
                   const float* __restrict__ b_p, float* __restrict__ off) {
  __shared__ float wp[NOFF * CIN * 9];  // 10368 floats
  int tid = threadIdx.x;
  for (int idx = tid; idx < NOFF * CIN * 9; idx += 256) wp[idx] = w_p[idx];
  __syncthreads();

  int g = blockIdx.x * 256 + tid;       // 0 .. 131071
  int b = g >> 14;
  int i = (g >> 7) & 127;
  int j = g & 127;

  float acc[NOFF];
#pragma unroll
  for (int oc = 0; oc < NOFF; ++oc) acc[oc] = b_p[oc];

  const float* xb = x + (size_t)b * CIN * HWSZ;
  for (int c = 0; c < CIN; ++c) {
    const float* xc = xb + c * HWSZ;
#pragma unroll
    for (int kh = 0; kh < 3; ++kh) {
      int ii = i + kh - 1;
      bool okr = (unsigned)ii < 128u;
#pragma unroll
      for (int kw = 0; kw < 3; ++kw) {
        int jj = j + kw - 1;
        float xv = (okr && ((unsigned)jj < 128u)) ? xc[ii * WW + jj] : 0.f;
#pragma unroll
        for (int oc = 0; oc < NOFF; ++oc)
          acc[oc] = fmaf(xv, wp[((oc * CIN + c) * 3 + kh) * 3 + kw], acc[oc]);
      }
    }
  }
#pragma unroll
  for (int oc = 0; oc < NOFF; ++oc)
    off[((size_t)(b * NOFF + oc) * HH + i) * WW + j] = acc[oc];
}

// ---------------------------------------------------------------------------
// Kernel 2: repack w_conv[oc][c][tap] (f32) -> wt[tap][oc][c] (bf16) so that
// WMMA B-fragments are contiguous 32B per lane.
// ---------------------------------------------------------------------------
__global__ void __launch_bounds__(256)
weight_pack_kernel(const float* __restrict__ w_conv, unsigned short* __restrict__ wt) {
  int t = blockIdx.x * 256 + threadIdx.x;    // 36864 = 9*64*64
  int n  = t >> 12;        // tap
  int oc = (t >> 6) & 63;
  int c  = t & 63;
  wt[t] = f32_to_bf16(w_conv[((size_t)oc * 64 + c) * 9 + n]);
}

// ---------------------------------------------------------------------------
// Kernel 3: fused deformable-sampling + implicit GEMM via WMMA bf16.
// One workgroup (256 thr = 8 waves) per (b, i) row: M-tile = 128 pixels,
// N = 64 out channels, K = 9 taps x 64 ch, stepped K=32 (tap x half-channels).
// ---------------------------------------------------------------------------
__global__ void __launch_bounds__(256)
deform_gemm_kernel(const float* __restrict__ x, const float* __restrict__ off,
                   const unsigned short* __restrict__ wt, float* __restrict__ out) {
  __shared__ unsigned short __attribute__((aligned(32))) Atile[128 * 32]; // 8 KB bf16
  __shared__ int   sm_i[4][NTAP * 128];   // 4 neighbor offsets per (tap,pixel)
  __shared__ float sm_g[4][NTAP * 128];   // 4 bilinear weights (0 when OOB)

  int tid = threadIdx.x;
  int b = blockIdx.x >> 7;
  int i = blockIdx.x & 127;

  // -------- phase 0: sampling geometry for this row (9 taps x 128 pixels) ----
  for (int p = tid; p < NTAP * 128; p += 256) {
    int n = p >> 7;
    int m = p & 127;
    float offx = off[((size_t)(b * NOFF + n)     * HH + i) * WW + m];
    float offy = off[((size_t)(b * NOFF + 9 + n) * HH + i) * WW + m];
    // padded-image coords: p0 (starts at 1) + p_n + offset
    float px = (float)(i + (n / 3)) + offx;        // (i+1) + (n/3 - 1)
    float py = (float)(m + (n % 3)) + offy;        // (m+1) + (n%3 - 1)
    float flx = floorf(px), fly = floorf(py);
    float qltx = fminf(fmaxf(flx,       0.f), 129.f);
    float qlty = fminf(fmaxf(fly,       0.f), 129.f);
    float qrbx = fminf(fmaxf(flx + 1.f, 0.f), 129.f);
    float qrby = fminf(fmaxf(fly + 1.f, 0.f), 129.f);
    float pxc  = fminf(fmaxf(px,        0.f), 129.f);
    float pyc  = fminf(fmaxf(py,        0.f), 129.f);
    float gx_lt = 1.f + qltx - pxc, gx_rb = 1.f - qrbx + pxc;
    float gy_lt = 1.f + qlty - pyc, gy_rb = 1.f - qrby + pyc;
    float g[4]  = { gx_lt * gy_lt, gx_rb * gy_rb, gx_lt * gy_rb, gx_rb * gy_lt };
    float qx[4] = { qltx, qrbx, qltx, qrbx };
    float qy[4] = { qlty, qrby, qrby, qlty };
#pragma unroll
    for (int s = 0; s < 4; ++s) {
      int xi = (int)qx[s] - 1, yi = (int)qy[s] - 1;  // back to unpadded coords
      bool inb = ((unsigned)xi < 128u) && ((unsigned)yi < 128u);
      sm_i[s][p] = inb ? (xi * WW + yi) : 0;
      sm_g[s][p] = inb ? g[s] : 0.f;                 // pad region contributes 0
    }
  }

  int lane  = tid & 31;
  int wv    = tid >> 5;      // wave id = M-subtile (16 pixels)
  int lrow  = lane & 15;
  int khalf = lane >> 4;

  v8f acc[4];
#pragma unroll
  for (int t = 0; t < 4; ++t)
#pragma unroll
    for (int e = 0; e < 8; ++e) acc[t][e] = 0.f;

  const float* xb = x + (size_t)b * CIN * HWSZ;

  // -------- K loop: 18 steps of K=32 (tap n, channel half ch) ---------------
  for (int kt = 0; kt < 18; ++kt) {
    int n  = kt >> 1;
    int ch = (kt & 1) << 5;
    __syncthreads();   // protect Atile against previous iteration's readers
    // build bf16 A-tile [128 pixels][32 channels] via bilinear gather
#pragma unroll
    for (int it = 0; it < 16; ++it) {
      int s  = it * 256 + tid;       // 4096 samples
      int kc = s >> 7;
      int m  = s & 127;
      int p  = (n << 7) + m;
      const float* base = xb + (size_t)(ch + kc) * HWSZ;
      float v = sm_g[0][p] * base[sm_i[0][p]]
              + sm_g[1][p] * base[sm_i[1][p]]
              + sm_g[2][p] * base[sm_i[2][p]]
              + sm_g[3][p] * base[sm_i[3][p]];
      Atile[m * 32 + kc] = f32_to_bf16(v);
    }
    __syncthreads();

    // A fragment (16x32 bf16): lane half 0 holds K 0-15, half 1 K 16-31
    int arow = (wv * 16 + lrow) * 32;
    v8u au = *(const v8u*)&Atile[arow + khalf * 16];
    v16bf a = __builtin_bit_cast(v16bf, au);

#pragma unroll
    for (int t = 0; t < 4; ++t) {
      // B fragment (32x16 bf16): lanes 0-15 K=0..15, lanes 16-31 K=16..31,
      // column = lrow; wt layout [tap][oc][c] makes this one 32B contiguous read
      const unsigned short* bptr =
          wt + ((size_t)(n * 64 + t * 16 + lrow) * 64 + ch + khalf * 16);
      v8u bu = *(const v8u*)bptr;
      v16bf bf = __builtin_bit_cast(v16bf, bu);
      acc[t] = __builtin_amdgcn_wmma_f32_16x16x32_bf16(
          false, a, false, bf, (short)0, acc[t], false, false);
    }
  }

  // -------- epilogue: D layout lane<16 -> M=r, lane>=16 -> M=8+r; N=lane%16 --
#pragma unroll
  for (int t = 0; t < 4; ++t) {
    int oc = t * 16 + lrow;
    float* orow = out + ((size_t)(b * 64 + oc) * HH + i) * WW;
#pragma unroll
    for (int r = 0; r < 8; ++r) {
      int j = wv * 16 + khalf * 8 + r;
      orow[j] = acc[t][r];
    }
  }
}

// ---------------------------------------------------------------------------
extern "C" void kernel_launch(void* const* d_in, const int* in_sizes, int n_in,
                              void* d_out, int out_size, void* d_ws, size_t ws_size,
                              hipStream_t stream) {
  (void)in_sizes; (void)n_in; (void)out_size; (void)ws_size;
  const float* x      = (const float*)d_in[0];
  const float* w_p    = (const float*)d_in[1];
  const float* b_p    = (const float*)d_in[2];
  const float* w_conv = (const float*)d_in[3];
  float* out = (float*)d_out;

  float* d_off = (float*)d_ws;                                  // 8*18*128*128 f32 = 9.4 MB
  unsigned short* d_wt =
      (unsigned short*)((char*)d_ws + (size_t)BATCH * NOFF * HWSZ * sizeof(float));

  offset_conv_kernel<<<512, 256, 0, stream>>>(x, w_p, b_p, d_off);
  weight_pack_kernel<<<144, 256, 0, stream>>>(w_conv, d_wt);
  deform_gemm_kernel<<<BATCH * HH, 256, 0, stream>>>(x, d_off, d_wt, out);
}
